// AriaGroupedExpertsMLP_64544768524490
// MI455X (gfx1250) — compile-verified, matrix operand order checked by reading, not run on hardware
//
#include <hip/hip_runtime.h>

#define NUM_TOKENS 1024
#define HIDDEN     2048
#define INTER      4096
#define EXPERTS    8

#define BM   128            // token rows per tile (== tokens per expert)
#define BK   32             // K per WMMA step (bf16 16x16x32)
#define BN   64             // output cols per matrix per block
#define STRA 40             // LDS row stride in halves (80B: 16B aligned, bank-spread)
#define STRB 40

#define AS1 __attribute__((address_space(1)))

typedef __attribute__((ext_vector_type(16))) __bf16 v16bf;
typedef __attribute__((ext_vector_type(2)))  __bf16 v2bf;
typedef __attribute__((ext_vector_type(8)))  float  v8f;
typedef __attribute__((ext_vector_type(4)))  float  f32x4;
typedef __attribute__((ext_vector_type(2)))  float  f32x2;

__device__ __forceinline__ const AS1 float* gfloat(const void* p) {
    return (const AS1 float*)(uintptr_t)p;
}

// native f32 -> bf16 (v_cvt_pk_bf16_f32)
__device__ __forceinline__ unsigned short bfbits(float x) {
    return __builtin_bit_cast(unsigned short, (__bf16)x);
}
__device__ __forceinline__ unsigned bfpk(float x, float y) {
    v2bf h;
    h[0] = (__bf16)x;
    h[1] = (__bf16)y;
    return __builtin_bit_cast(unsigned, h);
}

// silu via fast reciprocal: p * v_rcp_f32(1 + exp(-p)) -- avoids IEEE div expansion
__device__ __forceinline__ float silu_fast(float p) {
    return p * __builtin_amdgcn_rcpf(1.0f + __expf(-p));
}

// Load a 16-element bf16 fragment from LDS (16B-aligned start guaranteed by STRA/STRB)
__device__ __forceinline__ v16bf ldfrag(const unsigned short* p) {
    union { uint4 q[2]; v16bf v; } u;
    u.q[0] = *(const uint4*)(p);
    u.q[1] = *(const uint4*)(p + 8);
    return u.v;
}

__device__ __forceinline__ v8f wmma_bf16(v16bf a, v16bf b, v8f c) {
    return __builtin_amdgcn_wmma_f32_16x16x32_bf16(
        /*neg_a=*/false, a, /*neg_b=*/false, b,
        /*c_mod=*/(short)0, c, /*reuse_a=*/false, /*reuse_b=*/false);
}

// async global->LDS copy, 16B per lane, tracked by ASYNCcnt (CDNA5 ISA 10.6/15.18)
__device__ __forceinline__ void async_copy_b128(unsigned lds_addr, unsigned long long ga) {
    asm volatile("global_load_async_to_lds_b128 %0, %1, off"
                 :
                 : "v"(lds_addr), "v"(ga)
                 : "memory");
}
__device__ __forceinline__ void wait_asynccnt0() {
    asm volatile("s_wait_asynccnt 0x0" ::: "memory");
}
__device__ __forceinline__ unsigned lds_addr_of(const void* p) {
    // low 32 bits of a flat LDS pointer are the LDS byte address (ISA 10.2 aperture map)
    return (unsigned)(uintptr_t)p;
}

// ---------------------------------------------------------------------------
// Kernel 1: grouped fc1 (proj + gate halves) fused with SwiGLU, bf16 hidden out
// grid = (INTER/BN, EXPERTS), block = 256 (8 waves)
// ---------------------------------------------------------------------------
__global__ __launch_bounds__(256) void moe_fc1_swiglu(
    const float* __restrict__ x,          // [NUM_TOKENS, HIDDEN]
    const float* __restrict__ w1,         // [E, HIDDEN, 2*INTER]
    const int*   __restrict__ tpe,        // [E]
    unsigned short* __restrict__ hidden)  // [NUM_TOKENS, INTER] bf16
{
    __shared__ unsigned short lsA [BM * STRA];   // 10 KB
    __shared__ unsigned short lsBp[BN * STRB];   // 5 KB
    __shared__ unsigned short lsBg[BN * STRB];   // 5 KB

    const int e  = blockIdx.y;
    const int nb = blockIdx.x * BN;

    int base = 0;
    for (int i = 0; i < e; ++i) base += tpe[i];   // scalar prefix sum
    const int rows = tpe[e];
    const int rmax = (rows > 0) ? rows - 1 : 0;

    const AS1 float* xg = gfloat(x  + (size_t)base * HIDDEN);
    const AS1 float* wg = gfloat(w1 + (size_t)e * HIDDEN * (2 * INTER));

    const int tid  = threadIdx.x;
    const int lane = tid & 31;
    const int wave = tid >> 5;
    const int wm   = (wave & 3) << 5;    // wave row offset: 0,32,64,96
    const int wn   = (wave >> 2) << 5;   // wave col offset: 0 or 32
    const int lm   = lane & 15;
    const int koff = (lane >> 4) << 4;   // 0 or 16

    // ---- A staging slots: 4x (row, 4 cols) per thread; row clamped so reads
    //      stay in-bounds when rows < BM (epilogue guards correctness).
    unsigned offA[4]; unsigned short* sa[4];
    #pragma unroll
    for (int j = 0; j < 4; ++j) {
        const int s  = tid + 256 * j;
        const int r  = s >> 3;            // 0..127 (LDS row)
        const int c4 = (s & 7) << 2;      // 0..28
        const int rc = (r < rows) ? r : rmax;  // clamped global row
        offA[j] = (unsigned)(rc * HIDDEN + c4);
        sa[j]   = &lsA[r * STRA + c4];
    }
    // ---- B staging: each thread owns a 4x4 micro-tile of one matrix (p or g),
    //      micro-transposed: 4 global f32x4 rows -> 4 ds_store_b64 columns.
    const int bmat = tid >> 7;            // 0: proj, 1: gate
    const int bs   = tid & 127;
    const int br   = (bs >> 4) << 2;      // k row group: 0,4,...,28
    const int bc   = (bs & 15) << 2;      // col group:   0,4,...,60
    unsigned offB = (unsigned)(br * (2 * INTER) + nb + bc + bmat * INTER);
    unsigned short* sB = (bmat ? lsBg : lsBp) + bc * STRB + br;

    const unsigned short* fA0 = &lsA [(wm +      lm) * STRA + koff];
    const unsigned short* fA1 = &lsA [(wm + 16 + lm) * STRA + koff];
    const unsigned short* fP0 = &lsBp[(wn +      lm) * STRB + koff];
    const unsigned short* fP1 = &lsBp[(wn + 16 + lm) * STRB + koff];
    const unsigned short* fG0 = &lsBg[(wn +      lm) * STRB + koff];
    const unsigned short* fG1 = &lsBg[(wn + 16 + lm) * STRB + koff];

    v8f acc_p[2][2] = {}, acc_g[2][2] = {};

    for (int kt = 0; kt < HIDDEN / BK; ++kt) {
        // ---- stage A: 128x32 f32 -> bf16 (packed cvt), 4 f32x4 per thread
        #pragma unroll
        for (int j = 0; j < 4; ++j) {
            const f32x4 v = *(const AS1 f32x4*)(xg + offA[j]);
            uint2 pk;
            pk.x = bfpk(v[0], v[1]);
            pk.y = bfpk(v[2], v[3]);
            *(uint2*)sa[j] = pk;
            offA[j] += BK;
        }
        // ---- stage B: 4x4 micro-transpose; weights streamed once -> NT hint
        {
            const f32x4 r0 = __builtin_nontemporal_load((const AS1 f32x4*)(wg + offB));
            const f32x4 r1 = __builtin_nontemporal_load((const AS1 f32x4*)(wg + offB + 2 * INTER));
            const f32x4 r2 = __builtin_nontemporal_load((const AS1 f32x4*)(wg + offB + 4 * INTER));
            const f32x4 r3 = __builtin_nontemporal_load((const AS1 f32x4*)(wg + offB + 6 * INTER));
            #pragma unroll
            for (int c = 0; c < 4; ++c) {
                uint2 col;
                col.x = bfpk(r0[c], r1[c]);
                col.y = bfpk(r2[c], r3[c]);
                *(uint2*)(sB + c * STRB) = col;
            }
            offB += BK * (2 * INTER);
        }
        __syncthreads();

        // ---- fragments + 8 WMMA (uniform control flow, EXEC all ones)
        const v16bf a0 = ldfrag(fA0);
        const v16bf a1 = ldfrag(fA1);
        const v16bf p0 = ldfrag(fP0);
        const v16bf p1 = ldfrag(fP1);
        const v16bf g0 = ldfrag(fG0);
        const v16bf g1 = ldfrag(fG1);

        acc_p[0][0] = wmma_bf16(a0, p0, acc_p[0][0]);
        acc_p[0][1] = wmma_bf16(a0, p1, acc_p[0][1]);
        acc_p[1][0] = wmma_bf16(a1, p0, acc_p[1][0]);
        acc_p[1][1] = wmma_bf16(a1, p1, acc_p[1][1]);
        acc_g[0][0] = wmma_bf16(a0, g0, acc_g[0][0]);
        acc_g[0][1] = wmma_bf16(a0, g1, acc_g[0][1]);
        acc_g[1][0] = wmma_bf16(a1, g0, acc_g[1][0]);
        acc_g[1][1] = wmma_bf16(a1, g1, acc_g[1][1]);
        __syncthreads();
    }

    // ---- epilogue: SwiGLU (rcp-based silu), write bf16 hidden
    // C/D layout: VGPR r, lane l -> M = r + 8*(l>=16), N = l & 15
    AS1 unsigned short* hg = (AS1 unsigned short*)(uintptr_t)hidden;
    const int mhi = (lane & 16) ? 8 : 0;
    if (rows == BM) {
        // fast path: single base address, row/col offsets fold into store ioffset
        AS1 unsigned short* hp =
            hg + (size_t)(base + wm + mhi) * INTER + nb + wn + lm;
        #pragma unroll
        for (int i = 0; i < 2; ++i)
            #pragma unroll
            for (int j = 0; j < 2; ++j)
                #pragma unroll
                for (int r = 0; r < 8; ++r)
                    hp[(16 * i + r) * INTER + 16 * j] =
                        bfbits(silu_fast(acc_p[i][j][r]) * acc_g[i][j][r]);
    } else {
        #pragma unroll
        for (int i = 0; i < 2; ++i)
            #pragma unroll
            for (int j = 0; j < 2; ++j)
                #pragma unroll
                for (int r = 0; r < 8; ++r) {
                    const int mrow = wm + 16 * i + mhi + r;
                    const int col  = wn + 16 * j + lm;
                    if (mrow < rows)
                        hg[(size_t)(base + mrow) * INTER + nb + col] =
                            bfbits(silu_fast(acc_p[i][j][r]) * acc_g[i][j][r]);
                }
    }
}

// ---------------------------------------------------------------------------
// Kernel 2: grouped fc2, bf16 hidden (A, async-staged) x f32->bf16 w2 (B)
// grid = (HIDDEN/BN, EXPERTS), block = 256 (8 waves)
// ---------------------------------------------------------------------------
__global__ __launch_bounds__(256) void moe_fc2(
    const unsigned short* __restrict__ hidden, // [NUM_TOKENS, INTER] bf16
    const float* __restrict__ w2,              // [E, INTER, HIDDEN]
    const int*   __restrict__ tpe,             // [E]
    float* __restrict__ out)                   // [NUM_TOKENS, HIDDEN]
{
    __shared__ unsigned short lsA[BM * STRA];
    __shared__ unsigned short lsB[BN * STRB];

    const int e  = blockIdx.y;
    const int nb = blockIdx.x * BN;

    int base = 0;
    for (int i = 0; i < e; ++i) base += tpe[i];
    const int rows = tpe[e];

    const unsigned short* ha = hidden + (size_t)base * INTER;
    const AS1 float* wg = gfloat(w2 + (size_t)e * INTER * HIDDEN);

    const int tid  = threadIdx.x;
    const int lane = tid & 31;
    const int wave = tid >> 5;
    const int wm   = (wave & 3) << 5;
    const int wn   = (wave >> 2) << 5;
    const int lm   = lane & 15;
    const int koff = (lane >> 4) << 4;

    // ---- A staging: async global->LDS, 2x 16B per thread (pure bf16 copy)
    unsigned long long gaA0, gaA1; unsigned ldsA0, ldsA1;
    {
        const int s0 = tid,       r0 = s0 >> 2, c80 = (s0 & 3) << 3;
        const int s1 = tid + 256, r1 = s1 >> 2, c81 = (s1 & 3) << 3;
        gaA0  = (unsigned long long)(uintptr_t)(ha + (size_t)r0 * INTER + c80);
        gaA1  = (unsigned long long)(uintptr_t)(ha + (size_t)r1 * INTER + c81);
        ldsA0 = lds_addr_of(&lsA[r0 * STRA + c80]);
        ldsA1 = lds_addr_of(&lsA[r1 * STRA + c81]);
    }
    // ---- B staging: 4x2 micro-transpose per thread
    const int br = (tid >> 5) << 2;       // k row group: 0..28
    const int bc = (tid & 31) << 1;       // col group:   0..62
    unsigned offB = (unsigned)(br * HIDDEN + nb + bc);
    unsigned short* sB = &lsB[bc * STRB + br];

    const unsigned short* fA0 = &lsA[(wm +      lm) * STRA + koff];
    const unsigned short* fA1 = &lsA[(wm + 16 + lm) * STRA + koff];
    const unsigned short* fB0 = &lsB[(wn +      lm) * STRB + koff];
    const unsigned short* fB1 = &lsB[(wn + 16 + lm) * STRB + koff];

    v8f acc[2][2] = {};

    for (int kt = 0; kt < INTER / BK; ++kt) {
        // ---- stage A: bf16 copy global->LDS via ASYNC path (no VGPR transit)
        async_copy_b128(ldsA0, gaA0);
        async_copy_b128(ldsA1, gaA1);
        gaA0 += BK * sizeof(unsigned short);
        gaA1 += BK * sizeof(unsigned short);

        // ---- stage B: f32 -> bf16 micro-transpose; weights streamed once -> NT
        {
            const f32x2 r0 = __builtin_nontemporal_load((const AS1 f32x2*)(wg + offB));
            const f32x2 r1 = __builtin_nontemporal_load((const AS1 f32x2*)(wg + offB + 1 * HIDDEN));
            const f32x2 r2 = __builtin_nontemporal_load((const AS1 f32x2*)(wg + offB + 2 * HIDDEN));
            const f32x2 r3 = __builtin_nontemporal_load((const AS1 f32x2*)(wg + offB + 3 * HIDDEN));
            #pragma unroll
            for (int c = 0; c < 2; ++c) {
                uint2 col;
                col.x = bfpk(r0[c], r1[c]);
                col.y = bfpk(r2[c], r3[c]);
                *(uint2*)(sB + c * STRB) = col;
            }
            offB += BK * HIDDEN;
        }

        wait_asynccnt0();      // this wave's async A-copies landed in LDS
        __syncthreads();       // all waves' staging visible

        const v16bf a0 = ldfrag(fA0);
        const v16bf a1 = ldfrag(fA1);
        const v16bf b0 = ldfrag(fB0);
        const v16bf b1 = ldfrag(fB1);

        acc[0][0] = wmma_bf16(a0, b0, acc[0][0]);
        acc[0][1] = wmma_bf16(a0, b1, acc[0][1]);
        acc[1][0] = wmma_bf16(a1, b0, acc[1][0]);
        acc[1][1] = wmma_bf16(a1, b1, acc[1][1]);
        __syncthreads();
    }

    AS1 float* og = (AS1 float*)(uintptr_t)out;
    const int mhi = (lane & 16) ? 8 : 0;
    if (rows == BM) {
        AS1 float* op = og + (size_t)(base + wm + mhi) * HIDDEN + nb + wn + lm;
        #pragma unroll
        for (int i = 0; i < 2; ++i)
            #pragma unroll
            for (int j = 0; j < 2; ++j)
                #pragma unroll
                for (int r = 0; r < 8; ++r)
                    op[(16 * i + r) * HIDDEN + 16 * j] = acc[i][j][r];
    } else {
        #pragma unroll
        for (int i = 0; i < 2; ++i)
            #pragma unroll
            for (int j = 0; j < 2; ++j)
                #pragma unroll
                for (int r = 0; r < 8; ++r) {
                    const int mrow = wm + 16 * i + mhi + r;
                    const int col  = wn + 16 * j + lm;
                    if (mrow < rows)
                        og[(size_t)(base + mrow) * HIDDEN + nb + col] = acc[i][j][r];
                }
    }
}

extern "C" void kernel_launch(void* const* d_in, const int* in_sizes, int n_in,
                              void* d_out, int out_size, void* d_ws, size_t ws_size,
                              hipStream_t stream) {
    const float* x   = (const float*)d_in[0];   // [1024, 2048]
    const float* w1  = (const float*)d_in[1];   // [8, 2048, 8192]
    const float* w2  = (const float*)d_in[2];   // [8, 4096, 2048]
    const int*   tpe = (const int*)d_in[3];     // [8]
    float* out = (float*)d_out;                 // [1024, 2048]

    unsigned short* hidden = (unsigned short*)d_ws;   // [1024, 4096] bf16 = 8 MB

    dim3 blk(256);
    dim3 g1(INTER / BN, EXPERTS);    // 64 x 8 = 512 blocks
    dim3 g2(HIDDEN / BN, EXPERTS);   // 32 x 8 = 256 blocks

    moe_fc1_swiglu<<<g1, blk, 0, stream>>>(x, w1, tpe, hidden);
    moe_fc2<<<g2, blk, 0, stream>>>(hidden, w2, tpe, out);
}
